// TokenFlow_30562987278579
// MI455X (gfx1250) — matrix operand order repaired and checked by reading, not compiled, not run on hardware
//
#include <hip/hip_runtime.h>
#include <hip/hip_bf16.h>

typedef _Float16 v8h  __attribute__((ext_vector_type(8)));
typedef _Float16 v16h __attribute__((ext_vector_type(16)));
typedef float    v8f  __attribute__((ext_vector_type(8)));

#define N_TOK 8192      // B*H*W = 32*16*16
#define NCH   32        // channels
#define KCB   16384     // codebook size
#define WAVES 8
#define TINV  100.0f    // 1/ENTROPY_TEMPERATURE

// ---------------- prep kernels ----------------

__global__ void k_init(float* avg_probs, float* scalars) {
    int i = blockIdx.x * blockDim.x + threadIdx.x;
    if (i < KCB) avg_probs[i] = 0.0f;
    if (i < 4)   scalars[i]   = 0.0f;
}

// z [B=32, C=32, H=16, W=16] -> zf16 [N][C] row-major f16 + zn2[N]
__global__ void k_prep_z(const float* __restrict__ z, _Float16* __restrict__ zf16,
                         float* __restrict__ zn2) {
    int n = blockIdx.x * blockDim.x + threadIdx.x;
    if (n >= N_TOK) return;
    int b = n >> 8, hw = n & 255;
    const float* base = z + (size_t)b * (NCH * 256) + hw;
    float acc = 0.0f;
    #pragma unroll
    for (int c = 0; c < NCH; ++c) {
        float v = base[c * 256];
        acc += v * v;
        zf16[(size_t)n * NCH + c] = (_Float16)v;
    }
    zn2[n] = acc;
}

// embedding [K][C] -> e16 [K][C] f16 + en2[K]
__global__ void k_prep_e(const float* __restrict__ emb, _Float16* __restrict__ e16,
                         float* __restrict__ en2) {
    int k = blockIdx.x * blockDim.x + threadIdx.x;
    if (k >= KCB) return;
    const float* row = emb + (size_t)k * NCH;
    float acc = 0.0f;
    #pragma unroll
    for (int c = 0; c < NCH; ++c) {
        float v = row[c];
        acc += v * v;
        e16[(size_t)k * NCH + c] = (_Float16)v;
    }
    en2[k] = acc;
}

// ---------- helpers: per-wave operand packing per ISA 7.12.2 layouts ----------
__device__ __forceinline__ v16h pack_A(const _Float16* zf16, int row_base, int r, int half) {
    const v8h* zr = (const v8h*)(zf16 + (size_t)(row_base + r) * NCH);
    v8h a0 = zr[half];       // half0: ch0-7   half1: ch8-15
    v8h a1 = zr[2 + half];   // half0: ch16-23 half1: ch24-31
    v16h A;
    #pragma unroll
    for (int i = 0; i < 8; ++i) { A[i] = a0[i]; A[8 + i] = a1[i]; }
    return A;
}

__device__ __forceinline__ v16h pack_B(const _Float16* e16, int code, int half) {
    const v8h* er = (const v8h*)(e16 + (size_t)code * NCH);
    v8h b0 = er[half];       // half0: ch0-7   half1: ch8-15
    v8h b1 = er[2 + half];   // half0: ch16-23 half1: ch24-31
    v16h B;
    #pragma unroll
    for (int i = 0; i < 8; ++i) { B[i] = b0[i]; B[8 + i] = b1[i]; }
    return B;
}

// ---------------- sweep 1: argmin via WMMA (branchless min/select) ----------------
// grid: N/16 blocks of 256 threads (8 waves); block owns 16 token rows.
__global__ __launch_bounds__(256) void k_assign(
    const _Float16* __restrict__ zf16, const _Float16* __restrict__ e16,
    const float* __restrict__ zn2, const float* __restrict__ en2,
    int* __restrict__ idxbuf, float* __restrict__ mrow) {

    int row_base = blockIdx.x * 16;
    int tid  = threadIdx.x;
    int wave = tid >> 5, lane = tid & 31;
    int r = lane & 15, half = lane >> 4;

    v16h A = pack_A(zf16, row_base, r, half);

    float zn2v[8];
    #pragma unroll
    for (int v = 0; v < 8; ++v) zn2v[v] = zn2[row_base + half * 8 + v];

    float bd[8]; int bi[8];
    #pragma unroll
    for (int v = 0; v < 8; ++v) { bd[v] = 3.402823466e38f; bi[v] = 0; }

    for (int cb = wave * 16; cb < KCB; cb += 16 * WAVES) {
        int code = cb + r;                       // lane's output column = code
        v16h Bv = pack_B(e16, code, half);
        v8f cacc = {};
        cacc = __builtin_amdgcn_wmma_f32_16x16x32_f16(false, A, false, Bv,
                                                      (short)0, cacc, false, false);
        float e2 = en2[code];
        #pragma unroll
        for (int v = 0; v < 8; ++v) {
            float d = zn2v[v] + e2 - 2.0f * cacc[v];
            bool lt = d < bd[v];                 // codes increase per lane -> first-min kept
            bd[v] = lt ? d : bd[v];
            bi[v] = lt ? code : bi[v];
        }
    }

    // Cross-lane/cross-wave min-reduce via LDS: [row 0..15][slot 0..127]
    __shared__ float sD[2048];
    __shared__ int   sI[2048];
    #pragma unroll
    for (int v = 0; v < 8; ++v) {
        int off = (8 * half + v) * 128 + wave * 16 + r;
        sD[off] = bd[v]; sI[off] = bi[v];
    }
    __syncthreads();

    if (tid < 16) {
        int row = tid;
        float d = 3.402823466e38f; int ii = 0;
        for (int sl = 0; sl < 128; ++sl) {
            int off = row * 128 + sl;
            float dd = sD[off]; int io = sI[off];
            bool take = (dd < d) || (dd == d && io < ii);
            d  = take ? dd : d;
            ii = take ? io : ii;
        }
        int n = row_base + row;
        idxbuf[n] = ii;
        mrow[n] = -TINV * d;   // exact softmax max: logits are -TINV*d, monotonic
    }
}

// ---------------- sweep 2: s = sum exp(logit-m), t = sum exp(logit-m)*logit ----------------
__global__ __launch_bounds__(256) void k_sumexp(
    const _Float16* __restrict__ zf16, const _Float16* __restrict__ e16,
    const float* __restrict__ zn2, const float* __restrict__ en2,
    const float* __restrict__ mrow, float* __restrict__ srow,
    float* __restrict__ scalars) {

    int row_base = blockIdx.x * 16;
    int tid  = threadIdx.x;
    int wave = tid >> 5, lane = tid & 31;
    int r = lane & 15, half = lane >> 4;

    v16h A = pack_A(zf16, row_base, r, half);

    float zn2v[8], mv[8];
    #pragma unroll
    for (int v = 0; v < 8; ++v) {
        int n = row_base + half * 8 + v;
        zn2v[v] = zn2[n];
        mv[v]   = mrow[n];
    }

    float sv[8], tv[8];
    #pragma unroll
    for (int v = 0; v < 8; ++v) { sv[v] = 0.0f; tv[v] = 0.0f; }

    for (int cb = wave * 16; cb < KCB; cb += 16 * WAVES) {
        int code = cb + r;
        v16h Bv = pack_B(e16, code, half);
        v8f cacc = {};
        cacc = __builtin_amdgcn_wmma_f32_16x16x32_f16(false, A, false, Bv,
                                                      (short)0, cacc, false, false);
        float e2 = en2[code];
        #pragma unroll
        for (int v = 0; v < 8; ++v) {
            float logit = -TINV * (zn2v[v] + e2 - 2.0f * cacc[v]);
            float e = __expf(logit - mv[v]);     // logit <= m guaranteed
            sv[v] += e;
            tv[v] += e * logit;
        }
    }

    __shared__ float sS[2048], sT[2048];
    #pragma unroll
    for (int v = 0; v < 8; ++v) {
        int off = (8 * half + v) * 128 + wave * 16 + r;
        sS[off] = sv[v]; sT[off] = tv[v];
    }
    __syncthreads();

    if (tid < 16) {
        int row = tid;
        float s = 0.0f, t = 0.0f;
        for (int sl = 0; sl < 128; ++sl) {
            int off = row * 128 + sl;
            s += sS[off];
            t += sT[off];
        }
        int n = row_base + row;
        srow[n] = s;
        float lse = mrow[n] + __logf(s);
        atomicAdd(&scalars[0], lse - t / s);     // sample-entropy numerator
    }
}

// ---------------- sweep 3: avg_probs accumulation (64KB LDS histogram) ----------------
// grid: N/128 blocks of 256 threads.
__global__ __launch_bounds__(256) void k_avgp(
    const _Float16* __restrict__ zf16, const _Float16* __restrict__ e16,
    const float* __restrict__ zn2, const float* __restrict__ en2,
    const float* __restrict__ mrow, const float* __restrict__ srow,
    float* __restrict__ avg_probs) {

    __shared__ float acc[KCB];
    int tid = threadIdx.x;
    for (int k = tid; k < KCB; k += 256) acc[k] = 0.0f;
    __syncthreads();

    int wave = tid >> 5, lane = tid & 31;
    int r = lane & 15, half = lane >> 4;

    for (int rt = 0; rt < 8; ++rt) {
        int row_base = blockIdx.x * 128 + rt * 16;

        v16h A = pack_A(zf16, row_base, r, half);

        float zn2v[8], mv[8], isv[8];
        #pragma unroll
        for (int v = 0; v < 8; ++v) {
            int n = row_base + half * 8 + v;
            zn2v[v] = zn2[n];
            mv[v]   = mrow[n];
            isv[v]  = 1.0f / srow[n];
        }

        for (int cb = wave * 16; cb < KCB; cb += 16 * WAVES) {
            int code = cb + r;
            v16h Bv = pack_B(e16, code, half);
            v8f cacc = {};
            cacc = __builtin_amdgcn_wmma_f32_16x16x32_f16(false, A, false, Bv,
                                                          (short)0, cacc, false, false);
            float e2 = en2[code];
            float psum = 0.0f;
            #pragma unroll
            for (int v = 0; v < 8; ++v) {
                float logit = -TINV * (zn2v[v] + e2 - 2.0f * cacc[v]);
                psum += __expf(logit - mv[v]) * isv[v];
            }
            atomicAdd(&acc[code], psum);         // ds_add_f32
        }
    }
    __syncthreads();
    for (int k = tid; k < KCB; k += 256) atomicAdd(&avg_probs[k], acc[k]);
}

// ---------------- avg entropy reduce ----------------
__global__ void k_avg_ent(const float* __restrict__ avg_probs, float* __restrict__ scalars) {
    __shared__ float partial[256];
    float local = 0.0f;
    for (int k = threadIdx.x; k < KCB; k += 256) {
        float q = avg_probs[k] * (1.0f / (float)N_TOK);
        local += q * __logf(q + 1e-5f);
    }
    partial[threadIdx.x] = local;
    __syncthreads();
    for (int s = 128; s > 0; s >>= 1) {
        if (threadIdx.x < (unsigned)s) partial[threadIdx.x] += partial[threadIdx.x + s];
        __syncthreads();
    }
    if (threadIdx.x == 0) scalars[2] = -partial[0];  // avg_entropy
}

// ---------------- gather + exact MSE + outputs ----------------
__global__ void k_final(const float* __restrict__ z, const float* __restrict__ emb,
                        const int* __restrict__ idxbuf, float* __restrict__ out,
                        float* __restrict__ scalars) {
    int n = blockIdx.x * blockDim.x + threadIdx.x;
    if (n >= N_TOK) return;
    int b = n >> 8, hw = n & 255;
    int id = idxbuf[n];
    const float* e  = emb + (size_t)id * NCH;
    const float* zb = z   + (size_t)b * (NCH * 256) + hw;
    float*       ob = out + (size_t)b * (NCH * 256) + hw;
    float mse = 0.0f;
    #pragma unroll
    for (int c = 0; c < NCH; ++c) {
        float ev = e[c], zv = zb[c * 256];
        float df = ev - zv;
        mse += df * df;
        ob[c * 256] = ev;                 // z_q_out (straight-through == z_q forward)
    }
    atomicAdd(&scalars[1], mse);
    out[32769 + n] = (float)id;           // idx output region
}

__global__ void k_loss(const float* __restrict__ scalars, float* __restrict__ out) {
    if (threadIdx.x == 0) {
        float mse_mean   = scalars[1] / (float)(N_TOK * NCH);
        float sample_ent = scalars[0] / (float)N_TOK;
        float ent        = sample_ent - scalars[2];
        // codebook + 0.25*commit (numerically equal) + 0.1*entropy
        out[32768] = 1.25f * mse_mean + 0.1f * ent;
    }
}

// ---------------- host launcher ----------------
extern "C" void kernel_launch(void* const* d_in, const int* in_sizes, int n_in,
                              void* d_out, int out_size, void* d_ws, size_t ws_size,
                              hipStream_t stream) {
    const float* z   = (const float*)d_in[0];   // [32,32,16,16]
    const float* emb = (const float*)d_in[1];   // [16384,32]
    float* out = (float*)d_out;                 // 32768 + 1 + 8192

    char* ws = (char*)d_ws;
    size_t off = 0;
    _Float16* zf16 = (_Float16*)(ws + off); off += (size_t)N_TOK * NCH * 2;  // 512KB
    _Float16* e16  = (_Float16*)(ws + off); off += (size_t)KCB   * NCH * 2;  // 1MB
    float* zn2       = (float*)(ws + off); off += N_TOK * 4;
    float* en2       = (float*)(ws + off); off += KCB   * 4;
    float* mrow      = (float*)(ws + off); off += N_TOK * 4;
    float* srow      = (float*)(ws + off); off += N_TOK * 4;
    int*   idxbuf    = (int*)  (ws + off); off += N_TOK * 4;
    float* avg_probs = (float*)(ws + off); off += KCB   * 4;
    float* scalars   = (float*)(ws + off); off += 16;   // [0]=SE sum [1]=MSE sum [2]=avg_ent

    k_init  <<<(KCB + 255) / 256, 256, 0, stream>>>(avg_probs, scalars);
    k_prep_z<<<N_TOK / 256, 256, 0, stream>>>(z, zf16, zn2);
    k_prep_e<<<KCB   / 256, 256, 0, stream>>>(emb, e16, en2);
    k_assign<<<N_TOK / 16,  256, 0, stream>>>(zf16, e16, zn2, en2, idxbuf, mrow);
    k_sumexp<<<N_TOK / 16,  256, 0, stream>>>(zf16, e16, zn2, en2, mrow, srow, scalars);
    k_avgp  <<<N_TOK / 128, 256, 0, stream>>>(zf16, e16, zn2, en2, mrow, srow, avg_probs);
    k_avg_ent<<<1, 256, 0, stream>>>(avg_probs, scalars);
    k_final <<<N_TOK / 256, 256, 0, stream>>>(z, emb, idxbuf, out, scalars);
    k_loss  <<<1, 64, 0, stream>>>(scalars, out);
}
